// LSTM_Cell_188978561241
// MI455X (gfx1250) — matrix operand order repaired and checked by reading, not compiled, not run on hardware
//
#include <hip/hip_runtime.h>
#include <hip/hip_bf16.h>

typedef __attribute__((ext_vector_type(16))) __bf16 v16bf;
typedef __attribute__((ext_vector_type(8)))  __bf16 v8bf;
typedef __attribute__((ext_vector_type(4)))  __bf16 v4bf;
typedef __attribute__((ext_vector_type(8)))  float  v8f;
typedef unsigned int u32;
typedef __attribute__((ext_vector_type(4))) u32 u32x4;
typedef __attribute__((ext_vector_type(2))) u32 u32x2;
typedef __attribute__((ext_vector_type(4))) int i32x4;
typedef __attribute__((ext_vector_type(8))) int i32x8;

#define BATCH   4096
#define HID     1024
#define KTOT    2048
#define KB      32
#define NK      (KTOT / KB)
#define KST     40      // LDS row stride in bf16 (80B, conflict-free, 16B aligned)
#define BM      128     // WG tile M
#define BN      32      // WG tile N (per gate)
#define WTM     32      // per-wave M tile

#define ACAT_ELEMS ((size_t)BATCH * KTOT)          // 8M bf16
#define WCAT_ELEMS ((size_t)4 * HID * KTOT)        // 8M bf16
#define WS_NEEDED  ((ACAT_ELEMS + WCAT_ELEMS) * 2) // 32 MB

#if __has_builtin(__builtin_amdgcn_tensor_load_to_lds)
#define HAVE_TDM 1
#else
#define HAVE_TDM 0
#endif

struct LstmParams {
    const float* x;
    const float* h1;
    const float* c1;
    const float* Wx[4];   // gate order i,f,o,c
    const float* Wh[4];
    const float* bx[4];
    const float* bh[4];
    float* outh;
    float* outc;
    __bf16* Acat;         // [4096][2048] bf16   (x | h)
    __bf16* Wcat;         // [4][1024][2048] bf16 (Wx | Wh per gate)
};

// ---- fp32 -> bf16, round-to-nearest via +0x8000, packed with v_perm_b32 ----
__device__ __forceinline__ v4bf pack4(float4 v) {
    u32 x0 = __builtin_bit_cast(u32, v.x) + 0x8000u;
    u32 x1 = __builtin_bit_cast(u32, v.y) + 0x8000u;
    u32 x2 = __builtin_bit_cast(u32, v.z) + 0x8000u;
    u32 x3 = __builtin_bit_cast(u32, v.w) + 0x8000u;
    u32x2 r;
    r[0] = __builtin_amdgcn_perm(x1, x0, 0x07060302u); // {x1.hi16, x0.hi16}
    r[1] = __builtin_amdgcn_perm(x3, x2, 0x07060302u);
    return __builtin_bit_cast(v4bf, r);
}

__device__ __forceinline__ float fsigmoid(float x) {
    return 1.0f / (1.0f + __expf(-x));
}
__device__ __forceinline__ float ftanh(float x) {
    float e = __expf(2.0f * x);
    return 1.0f - 2.0f / (e + 1.0f);
}

// =====================================================================
// Kernel 1: one-shot fp32 -> bf16 conversion into workspace.
// =====================================================================
__global__ __launch_bounds__(256) void lstm_convert(LstmParams p) {
    const size_t idx = (size_t)blockIdx.x * 256 + threadIdx.x; // one float4 each
    const size_t AV4 = ACAT_ELEMS / 4;                         // 2M
    float4 s;
    __bf16* dst;
    if (idx < AV4) {
        const int row = (int)(idx >> 9);
        const int col = ((int)idx & 511) * 4;
        const float* src = (col < HID) ? (p.x  + (size_t)row * HID + col)
                                       : (p.h1 + (size_t)row * HID + (col - HID));
        s = *(const float4*)src;
        dst = p.Acat + idx * 4;
    } else {
        const size_t j = idx - AV4;
        const int g   = (int)(j >> 19);
        const int rem = (int)(j & ((1u << 19) - 1));
        const int n   = rem >> 9;
        const int col = (rem & 511) * 4;
        const float* W = (col < HID) ? p.Wx[g] : p.Wh[g];
        s = *(const float4*)(W + (size_t)n * HID + (col & (HID - 1)));
        dst = p.Wcat + j * 4;
    }
    *(v4bf*)dst = pack4(s);
}

// =====================================================================
// TDM descriptor builders (all-uniform values; live in SGPRs)
// =====================================================================
__device__ __forceinline__ u32x4 tdm_g0(u32 lds_addr, unsigned long long gaddr) {
    u32x4 g;
    g[0] = 1u;                                    // count=1, no gather
    g[1] = lds_addr;                              // LDS byte address
    g[2] = (u32)gaddr;                            // global addr [31:0]
    g[3] = (u32)(gaddr >> 32) | (2u << 30);       // global addr [56:32] | type=2
    return g;
}

__device__ __forceinline__ i32x8 tdm_g1(u32 tile0, u32 tile1, u32 tile2,
                                        u32 td0, u32 td1,
                                        unsigned long long s0,
                                        unsigned long long s1) {
    // data_size=1 (2B), pad_enable=1, pad_interval=3 (16 DW), pad_amount=3 (4 DW)
    // -> LDS row stride = 64B data + 16B pad = 80B = KST bf16
    i32x8 g;
    g[0] = (int)((1u << 16) | (1u << 20) | (3u << 22) | (3u << 25));
    g[1] = (int)((td0 & 0xFFFFu) << 16);
    g[2] = (int)(((td0 >> 16) & 0xFFFFu) | ((td1 & 0xFFFFu) << 16));
    g[3] = (int)(((td1 >> 16) & 0xFFFFu) | (tile0 << 16));
    g[4] = (int)((tile1 & 0xFFFFu) | (tile2 << 16));
    g[5] = (int)(u32)(s0 & 0xFFFFFFFFull);
    g[6] = (int)(((u32)(s0 >> 32) & 0xFFFFu) | (((u32)(s1 & 0xFFFFull)) << 16));
    g[7] = (int)(u32)(s1 >> 16);
    return g;
}

#if HAVE_TDM
__device__ __forceinline__ void tdm_issue(u32x4 g0, i32x8 g1, i32x4 g2) {
#if __has_include(<hip/amd_detail/amd_gfx1250_TDM.h>)
    __builtin_amdgcn_tensor_load_to_lds(g0, g1, g2, (i32x4)0, (i32x8)0, 0);
#else
    __builtin_amdgcn_tensor_load_to_lds(g0, g1, g2, (i32x4)0, 0);
#endif
}
#endif

// =====================================================================
// Kernel 2: bf16 WMMA GEMM, TDM-fed double-buffered LDS, fused LSTM epilogue
// =====================================================================
__global__ __launch_bounds__(256) void lstm_gemm_tdm(LstmParams p) {
    __shared__ __bf16 As[2][BM * KST];       // 2 x 10240 B
    __shared__ __bf16 Bs[2][4 * BN * KST];   // 2 x 10240 B

    const int tid  = threadIdx.x;
    const int lane = tid & 31;
    const int wv   = tid >> 5;
    const int wm   = wv & 3;
    const int wn   = wv >> 2;
    const int Mbase = blockIdx.x * BM;
    const int Nbase = blockIdx.y * BN;

    v8f acc[4][2] = {};

    const int arow0 = wm * WTM + (lane & 15);
    const int abase = (lane >> 4) * 8;
    const int bnrow = wn * 16 + (lane & 15);
    const int bbase = (lane >> 4) * 8;

#if HAVE_TDM
    // A: 2D tile 32(k) x 128(m); B: 3D tile 32(k) x 32(n) x 4(gates)
    const i32x8 g1A = tdm_g1(32, 128, 0, 32, 128, KTOT, 0);
    const i32x8 g1B = tdm_g1(32, 32, 4, 32, 32, KTOT, (unsigned long long)HID * KTOT);
    const i32x4 g2A = (i32x4)0;
    i32x4 g2B = (i32x4)0; g2B[0] = 4;        // tensor_dim2 = 4 gates

    const unsigned long long gaA =
        (unsigned long long)(uintptr_t)(p.Acat + (size_t)Mbase * KTOT);
    const unsigned long long gaB =
        (unsigned long long)(uintptr_t)(p.Wcat + (size_t)Nbase * KTOT);
    const u32 ldsA[2] = { (u32)(uintptr_t)&As[0][0], (u32)(uintptr_t)&As[1][0] };
    const u32 ldsB[2] = { (u32)(uintptr_t)&Bs[0][0], (u32)(uintptr_t)&Bs[1][0] };

    if (wv == 0) {
        tdm_issue(tdm_g0(ldsA[0], gaA), g1A, g2A);
        tdm_issue(tdm_g0(ldsB[0], gaB), g1B, g2B);
        __builtin_amdgcn_s_wait_tensorcnt(0);
    }
    __syncthreads();
#else
    // cooperative bf16 copy fallback (no conversion: data pre-converted)
    const int cRow = tid >> 1;
    const int cSeg = (tid & 1) * 16;
    auto stage = [&](int buf, int kk) {
        const int k0 = kk * KB;
        {
            const __bf16* s = p.Acat + (size_t)(Mbase + cRow) * KTOT + k0 + cSeg;
            *(v8bf*)&As[buf][cRow * KST + cSeg]     = *(const v8bf*)s;
            *(v8bf*)&As[buf][cRow * KST + cSeg + 8] = *(const v8bf*)(s + 8);
        }
        {
            const int g = tid >> 6, n = (tid >> 1) & 31;
            const __bf16* s = p.Wcat + (size_t)((g << 10) + Nbase + n) * KTOT + k0 + cSeg;
            *(v8bf*)&Bs[buf][(g * BN + n) * KST + cSeg]     = *(const v8bf*)s;
            *(v8bf*)&Bs[buf][(g * BN + n) * KST + cSeg + 8] = *(const v8bf*)(s + 8);
        }
    };
    stage(0, 0);
    __syncthreads();
#endif

    for (int kk = 0; kk < NK; ++kk) {
        const int cur = kk & 1;

#if HAVE_TDM
        if (wv == 0 && kk + 1 < NK) {      // DMA next chunk under the WMMAs
            const unsigned long long off = (unsigned long long)(kk + 1) * KB * 2;
            tdm_issue(tdm_g0(ldsA[cur ^ 1], gaA + off), g1A, g2A);
            tdm_issue(tdm_g0(ldsB[cur ^ 1], gaB + off), g1B, g2B);
        }
#endif

        // ---- load ALL fragments first (distinct registers) so the DS
        //      loads pipeline ahead of the WMMA block ----
        v16bf af[2];
        #pragma unroll
        for (int t = 0; t < 2; ++t) {
            const int r = (arow0 + t * 16) * KST;
            v8bf alo = *(const v8bf*)&As[cur][r + abase];
            v8bf ahi = *(const v8bf*)&As[cur][r + abase + 16];
            af[t] = __builtin_shufflevector(alo, ahi,
                        0,1,2,3,4,5,6,7,8,9,10,11,12,13,14,15);
        }
        v16bf bf[4];
        #pragma unroll
        for (int g = 0; g < 4; ++g) {
            const int r = (g * BN + bnrow) * KST;
            v8bf blo = *(const v8bf*)&Bs[cur][r + bbase];
            v8bf bhi = *(const v8bf*)&Bs[cur][r + bbase + 16];
            bf[g] = __builtin_shufflevector(blo, bhi,
                        0,1,2,3,4,5,6,7,8,9,10,11,12,13,14,15);
        }

        #pragma unroll
        for (int g = 0; g < 4; ++g) {
            #pragma unroll
            for (int t = 0; t < 2; ++t) {
                acc[g][t] = __builtin_amdgcn_wmma_f32_16x16x32_bf16(
                                false, af[t], false, bf[g],
                                (short)0, acc[g][t], false, false);
            }
        }

#if HAVE_TDM
        if (wv == 0) __builtin_amdgcn_s_wait_tensorcnt(0);
#else
        if (kk + 1 < NK) stage(cur ^ 1, kk + 1);
#endif
        __syncthreads();
    }

    // ---- fused LSTM epilogue ----
    const int nIdx = Nbase + wn * 16 + (lane & 15);
    const int mTop = Mbase + wm * WTM + ((lane >> 4) << 3);

    const float bs0 = p.bx[0][nIdx] + p.bh[0][nIdx];
    const float bs1 = p.bx[1][nIdx] + p.bh[1][nIdx];
    const float bs2 = p.bx[2][nIdx] + p.bh[2][nIdx];
    const float bs3 = p.bx[3][nIdx] + p.bh[3][nIdx];

    #pragma unroll
    for (int t = 0; t < 2; ++t) {
        #pragma unroll
        for (int r = 0; r < 8; ++r) {
            const int m = mTop + t * 16 + r;
            float iv = fsigmoid(acc[0][t][r] + bs0);
            float fv = fsigmoid(acc[1][t][r] + bs1);
            float ov = fsigmoid(acc[2][t][r] + bs2);
            float cv = ftanh   (acc[3][t][r] + bs3);
            float cp = p.c1[(size_t)m * HID + nIdx];
            float cn = fv * cp + iv * cv;
            float hn = ov * ftanh(cn);
            p.outh[(size_t)m * HID + nIdx] = hn;
            p.outc[(size_t)m * HID + nIdx] = cn;
        }
    }
}

// =====================================================================
// Fallback (no/undersized workspace): fused convert-in-loop kernel
// =====================================================================
__global__ __launch_bounds__(256) void lstm_fused(LstmParams p) {
    __shared__ __bf16 As[2][BM * KST];
    __shared__ __bf16 Bs[2][4 * BN * KST];

    const int tid  = threadIdx.x;
    const int lane = tid & 31;
    const int wv   = tid >> 5;
    const int wm   = wv & 3;
    const int wn   = wv >> 2;
    const int Mbase = blockIdx.x * BM;
    const int Nbase = blockIdx.y * BN;

    v8f acc[4][2] = {};

    const int ldRow = tid >> 3;
    const int ldCg  = tid & 7;
    const int arow0 = wm * WTM + (lane & 15);
    const int abase = (lane >> 4) * 8;
    const int bnrow = wn * 16 + (lane & 15);
    const int bbase = (lane >> 4) * 8;

    float4 aV[4], bV[4];

    auto fetch = [&](int kk) {
        const int k0 = kk * KB;
        const bool fh = (k0 < HID);
        const int ko = fh ? k0 : (k0 - HID);
        const float* Ag = fh ? (p.x + k0) : (p.h1 + ko);
        #pragma unroll
        for (int j = 0; j < 4; ++j)
            aV[j] = *(const float4*)(Ag + (size_t)(Mbase + ldRow + j * 32) * HID + ldCg * 4);
        #pragma unroll
        for (int g = 0; g < 4; ++g) {
            const float* Wg = fh ? p.Wx[g] : p.Wh[g];
            bV[g] = *(const float4*)(Wg + (size_t)(Nbase + ldRow) * HID + ko + ldCg * 4);
        }
    };
    auto stage = [&](int buf) {
        #pragma unroll
        for (int j = 0; j < 4; ++j)
            *(v4bf*)&As[buf][(ldRow + j * 32) * KST + ldCg * 4] = pack4(aV[j]);
        #pragma unroll
        for (int g = 0; g < 4; ++g)
            *(v4bf*)&Bs[buf][(g * BN + ldRow) * KST + ldCg * 4] = pack4(bV[g]);
    };

    fetch(0); stage(0);
    __syncthreads();

    for (int kk = 0; kk < NK; ++kk) {
        const int cur = kk & 1;
        if (kk + 1 < NK) fetch(kk + 1);

        v16bf af[2];
        #pragma unroll
        for (int t = 0; t < 2; ++t) {
            const int r = (arow0 + t * 16) * KST;
            v8bf alo = *(const v8bf*)&As[cur][r + abase];
            v8bf ahi = *(const v8bf*)&As[cur][r + abase + 16];
            af[t] = __builtin_shufflevector(alo, ahi,
                        0,1,2,3,4,5,6,7,8,9,10,11,12,13,14,15);
        }
        v16bf bf[4];
        #pragma unroll
        for (int g = 0; g < 4; ++g) {
            const int r = (g * BN + bnrow) * KST;
            v8bf blo = *(const v8bf*)&Bs[cur][r + bbase];
            v8bf bhi = *(const v8bf*)&Bs[cur][r + bbase + 16];
            bf[g] = __builtin_shufflevector(blo, bhi,
                        0,1,2,3,4,5,6,7,8,9,10,11,12,13,14,15);
        }
        #pragma unroll
        for (int g = 0; g < 4; ++g) {
            #pragma unroll
            for (int t = 0; t < 2; ++t) {
                acc[g][t] = __builtin_amdgcn_wmma_f32_16x16x32_bf16(
                                false, af[t], false, bf[g],
                                (short)0, acc[g][t], false, false);
            }
        }
        if (kk + 1 < NK) stage(cur ^ 1);
        __syncthreads();
    }

    const int nIdx = Nbase + wn * 16 + (lane & 15);
    const int mTop = Mbase + wm * WTM + ((lane >> 4) << 3);
    const float bs0 = p.bx[0][nIdx] + p.bh[0][nIdx];
    const float bs1 = p.bx[1][nIdx] + p.bh[1][nIdx];
    const float bs2 = p.bx[2][nIdx] + p.bh[2][nIdx];
    const float bs3 = p.bx[3][nIdx] + p.bh[3][nIdx];
    #pragma unroll
    for (int t = 0; t < 2; ++t) {
        #pragma unroll
        for (int r = 0; r < 8; ++r) {
            const int m = mTop + t * 16 + r;
            float iv = fsigmoid(acc[0][t][r] + bs0);
            float fv = fsigmoid(acc[1][t][r] + bs1);
            float ov = fsigmoid(acc[2][t][r] + bs2);
            float cv = ftanh   (acc[3][t][r] + bs3);
            float cp = p.c1[(size_t)m * HID + nIdx];
            float cn = fv * cp + iv * cv;
            float hn = ov * ftanh(cn);
            p.outh[(size_t)m * HID + nIdx] = hn;
            p.outc[(size_t)m * HID + nIdx] = cn;
        }
    }
}

extern "C" void kernel_launch(void* const* d_in, const int* in_sizes, int n_in,
                              void* d_out, int out_size, void* d_ws, size_t ws_size,
                              hipStream_t stream) {
    (void)in_sizes; (void)n_in; (void)out_size;

    LstmParams p;
    p.x  = (const float*)d_in[0];
    p.h1 = (const float*)d_in[1];
    p.c1 = (const float*)d_in[2];
    const int wx[4] = {3, 7, 11, 15};
    const int wh[4] = {5, 9, 13, 17};
    const int bx[4] = {4, 8, 12, 16};
    const int bh[4] = {6, 10, 14, 18};
    for (int g = 0; g < 4; ++g) {
        p.Wx[g] = (const float*)d_in[wx[g]];
        p.Wh[g] = (const float*)d_in[wh[g]];
        p.bx[g] = (const float*)d_in[bx[g]];
        p.bh[g] = (const float*)d_in[bh[g]];
    }
    p.outh = (float*)d_out;
    p.outc = p.outh + (size_t)BATCH * HID;
    p.Acat = (__bf16*)d_ws;
    p.Wcat = (__bf16*)d_ws + ACAT_ELEMS;

    dim3 block(256);
    if (ws_size >= WS_NEEDED) {
        const size_t totalV4 = (ACAT_ELEMS + WCAT_ELEMS) / 4;   // 4M
        lstm_convert<<<dim3((unsigned)(totalV4 / 256)), block, 0, stream>>>(p);
        lstm_gemm_tdm<<<dim3(BATCH / BM, HID / BN), block, 0, stream>>>(p);
    } else {
        lstm_fused<<<dim3(BATCH / BM, HID / BN), block, 0, stream>>>(p);
    }
}
